// BahdanauAttnDecoderRNN_withtopic_77532749627888
// MI455X (gfx1250) — matrix operand order repaired and checked by reading, not compiled, not run on hardware
//
#include <hip/hip_runtime.h>
#include <math.h>

// ---------------------------------------------------------------------------
// CDNA5 / gfx1250: wave32, WMMA bf16 16x16x32 (f32 accumulate)
// ---------------------------------------------------------------------------
typedef __bf16 bf16;
typedef __attribute__((ext_vector_type(16))) __bf16 v16bf;
typedef __attribute__((ext_vector_type(8)))  float  v8f;
typedef __attribute__((ext_vector_type(4)))  float  v4f;   // native vec for NT builtins

// ---- Fragment loaders (contiguous K per lane half, wave32) -------------------
// A: 16x32 bf16 tile, row-major in LDS with row stride ldk (elements).
// lane 0-15: M=lane, K=0..15; lane 16-31: M=lane-16, K=16..31
// -> one contiguous 16-element (32B) run per lane.
__device__ __forceinline__ v16bf lds_load_a_frag(const bf16* tile, int ldk, int k0) {
  int lane = threadIdx.x & 31;
  int m  = lane & 15;
  int kh = (lane >> 4) & 1;
  const bf16* row = tile + m * ldk + k0 + kh * 16;
  v16bf a;
#pragma unroll
  for (int v = 0; v < 8; ++v) {
    int kb = v * 2;   // 0,2,4,...,14
    a[2 * v]     = row[kb];
    a[2 * v + 1] = row[kb + 1];
  }
  return a;
}

// B: 32x16 (KxN) from row-major fp32 W[N][K], converted inline to bf16.
// lane 0-15: N=lane, K=0..15; lane 16-31: N=lane-16, K=16..31.
// Streamed weights are single-use -> non-temporal (keep enc resident in L2).
__device__ __forceinline__ v16bf g_load_b_frag_nt(const float* W, int ldw, int n0, int k0) {
  int lane = threadIdx.x & 31;
  int n  = lane & 15;
  int kh = (lane >> 4) & 1;
  const v4f* p = (const v4f*)(W + (size_t)(n0 + n) * ldw + k0 + kh * 16);
  v4f f0 = __builtin_nontemporal_load(p + 0);
  v4f f1 = __builtin_nontemporal_load(p + 1);
  v4f f2 = __builtin_nontemporal_load(p + 2);
  v4f f3 = __builtin_nontemporal_load(p + 3);
  v16bf b;
  b[0]=(bf16)f0.x;  b[1]=(bf16)f0.y;  b[2]=(bf16)f0.z;  b[3]=(bf16)f0.w;
  b[4]=(bf16)f1.x;  b[5]=(bf16)f1.y;  b[6]=(bf16)f1.z;  b[7]=(bf16)f1.w;
  b[8]=(bf16)f2.x;  b[9]=(bf16)f2.y;  b[10]=(bf16)f2.z; b[11]=(bf16)f2.w;
  b[12]=(bf16)f3.x; b[13]=(bf16)f3.y; b[14]=(bf16)f3.z; b[15]=(bf16)f3.w;
  return b;
}

// Guarded edge-tile loader: clamped addresses + mask-multiply, NO exec-mask
// branching (loads always execute in-bounds; invalid elements forced to 0).
__device__ __forceinline__ v16bf g_load_b_frag_guard(const float* W, int ldw,
                                                     int n0, int k0, int N, int K) {
  int lane = threadIdx.x & 31;
  int n  = lane & 15;
  int kh = (lane >> 4) & 1;
  int nn = n0 + n;
  float nmask = (nn < N) ? 1.0f : 0.0f;
  int rcl = (nn < N) ? nn : (N - 1);
  const float* row = W + (size_t)rcl * ldw;
  int kb = k0 + kh * 16;
  v16bf b;
#pragma unroll
  for (int i = 0; i < 16; ++i) {
    int k = kb + i;
    int kcl = (k < K) ? k : (K - 1);
    float msk = (k < K) ? nmask : 0.0f;
    b[i] = (bf16)(row[kcl] * msk);
  }
  return b;
}

// ---------------------------------------------------------------------------
// C[64 x N] = X[64 x K] @ W[N x K]^T + bias (+C if accumulate), bf16 WMMA.
// ONE block owns all 64 M-rows (4 accumulator tiles) -> each W element fetched
// from HBM exactly once. block = 128 thr (4 waves) -> 64x64 output tile.
// ---------------------------------------------------------------------------
#define GEMM_KC 128
#define GEMM_LDK 136   // 68 dwords/row -> conflict-free LDS row stride

__global__ void gemm_xwT_kernel(const float* __restrict__ X, int ldx,
                                const float* __restrict__ W, int ldw,
                                const float* __restrict__ bias,
                                float* __restrict__ C, int ldc,
                                int N, int K, int accumulate) {
  __shared__ bf16 atile[64 * GEMM_LDK];
  int tid  = threadIdx.x;
  int wave = tid >> 5;
  int lane = tid & 31;
  int n0 = blockIdx.x * 64 + wave * 16;

  v8f zero = {0.f,0.f,0.f,0.f,0.f,0.f,0.f,0.f};
  v8f acc[4];
  acc[0]=zero; acc[1]=zero; acc[2]=zero; acc[3]=zero;

  bool xal = ((ldx & 3) == 0);
  bool wal = ((ldw & 3) == 0);
  int nchunks = (K + GEMM_KC - 1) / GEMM_KC;
  for (int ch = 0; ch < nchunks; ++ch) {
    int kc = ch * GEMM_KC;
    bool fullk = (kc + GEMM_KC <= K);
    if (fullk && xal) {
      // full chunk: unguarded vec4 loads; pointer-increment addressing
      const float* xp = X + (size_t)wave * ldx + kc + lane * 4;
      size_t xstep = (size_t)4 * ldx;
      bf16* dp = atile + wave * GEMM_LDK + lane * 4;
#pragma unroll
      for (int it = 0; it < 16; ++it) {
        v4f f = *(const v4f*)xp;
        dp[0]=(bf16)f.x; dp[1]=(bf16)f.y; dp[2]=(bf16)f.z; dp[3]=(bf16)f.w;
        xp += xstep;
        dp += 4 * GEMM_LDK;
      }
    } else {
      // tail / unaligned chunk: clamped scalar loads + mask (no divergence)
      int kk  = kc + tid;
      int kcl = (kk < K) ? kk : (K - 1);
      float msk = (kk < K) ? 1.0f : 0.0f;
      const float* xp = X + kcl;
      bf16* dp = atile + tid;
#pragma unroll 4
      for (int it = 0; it < 64; ++it) {
        *dp = (bf16)(*xp * msk);
        xp += ldx;
        dp += GEMM_LDK;
      }
    }
    __syncthreads();
    bool fulln = wal && fullk && (n0 + 16 <= N);
#pragma unroll
    for (int ks = 0; ks < 4; ++ks) {
      v16bf b = fulln ? g_load_b_frag_nt(W, ldw, n0, kc + ks * 32)
                      : g_load_b_frag_guard(W, ldw, n0, kc + ks * 32, N, K);
#pragma unroll
      for (int mt = 0; mt < 4; ++mt) {
        v16bf a = lds_load_a_frag(atile + mt * 16 * GEMM_LDK, GEMM_LDK, ks * 32);
        acc[mt] = __builtin_amdgcn_wmma_f32_16x16x32_bf16(false, a, false, b,
                                                          (short)0, acc[mt], false, false);
      }
    }
    __syncthreads();
  }
  // C layout: lane 0-15 -> N=lane, M=j; lane 16-31 -> N=lane-16, M=8+j
  int n  = lane & 15;
  int mb = (lane >> 4) * 8;
  int nn = n0 + n;
  if (nn < N) {
    float bv = bias[nn];
#pragma unroll
    for (int mt = 0; mt < 4; ++mt) {
#pragma unroll
      for (int j = 0; j < 8; ++j) {
        int mm = mt * 16 + mb + j;        // < 64 always
        size_t off = (size_t)mm * ldc + nn;
        float v = acc[mt][j] + bv;
        if (accumulate) v += C[off];
        C[off] = v;
      }
    }
  }
}

// ---------------------------------------------------------------------------
// Fused keys + tanh + align: align[r] = sum_a tanh(q[b,a] + (enc_r . Wm_a))*Wal[a]
// r = s*64+b in [0,25600). Block: 256 thr (8 waves), 32 rows/block (800 blocks).
// enc staged once in LDS (bf16, 66KB, RT so it stays in L2 for the ctx pass);
// Wm pre-converted bf16 (L2-resident, B-frag = one 32B contiguous load).
// ---------------------------------------------------------------------------
#define KA_ROWS 32
#define KA_LDK 1032    // 516 dwords/row -> conflict-free row stride

__global__ void keys_align_kernel(const float* __restrict__ enc,
                                  const bf16* __restrict__ Wmb,
                                  const float* __restrict__ query,
                                  const float* __restrict__ Wal,
                                  float* __restrict__ align_out) {
  __shared__ bf16  atile[KA_ROWS * KA_LDK];
  __shared__ float align_s[KA_ROWS];
  int tid  = threadIdx.x;   // 0..255
  int wave = tid >> 5;      // 0..7
  int lane = tid & 31;
  int r0 = blockIdx.x * KA_ROWS;
  if (tid < KA_ROWS) align_s[tid] = 0.0f;
  // stage 32 rows x 1024 f32 -> bf16 (vec4, coalesced, one row per iter)
  {
    const float* ep = enc + (size_t)r0 * 1024 + tid * 4;
    bf16* dp = atile + tid * 4;
#pragma unroll
    for (int it = 0; it < KA_ROWS; ++it) {
      v4f f = *(const v4f*)ep;
      dp[0] = (bf16)f.x; dp[1] = (bf16)f.y; dp[2] = (bf16)f.z; dp[3] = (bf16)f.w;
      ep += 1024;
      dp += KA_LDK;
    }
  }
  __syncthreads();

  int n  = lane & 15;
  int kh = (lane >> 4) & 1;
  int mb = kh * 8;
  v8f zero = {0.f,0.f,0.f,0.f,0.f,0.f,0.f,0.f};
  v8f acc[2][4];
#pragma unroll
  for (int mt = 0; mt < 2; ++mt)
#pragma unroll
    for (int t = 0; t < 4; ++t) acc[mt][t] = zero;

  const bf16* wbase = Wmb + (size_t)(wave * 64 + n) * 1024 + kh * 16;
  for (int ks = 0; ks < 32; ++ks) {
    int k0 = ks * 32;
    v16bf bfrag[4];
#pragma unroll
    for (int t = 0; t < 4; ++t)    // 32B contiguous bf16 loads, batched
      bfrag[t] = *(const v16bf*)(wbase + (size_t)(t * 16) * 1024 + k0);
#pragma unroll
    for (int mt = 0; mt < 2; ++mt) {
      v16bf a = lds_load_a_frag(atile + mt * 16 * KA_LDK, KA_LDK, k0);
#pragma unroll
      for (int t = 0; t < 4; ++t)
        acc[mt][t] = __builtin_amdgcn_wmma_f32_16x16x32_bf16(false, a, false, bfrag[t],
                                                             (short)0, acc[mt][t], false, false);
    }
  }
  // tanh(query + keys) . W_align, reduced over this wave's 64 A-columns
  float pacc[2][8];
#pragma unroll
  for (int mt = 0; mt < 2; ++mt)
#pragma unroll
    for (int j = 0; j < 8; ++j) pacc[mt][j] = 0.0f;
#pragma unroll
  for (int t = 0; t < 4; ++t) {
    int nn = wave * 64 + t * 16 + n;
    float wv = Wal[nn];
#pragma unroll
    for (int mt = 0; mt < 2; ++mt) {
#pragma unroll
      for (int j = 0; j < 8; ++j) {
        int r = r0 + mt * 16 + mb + j;
        float kv = acc[mt][t][j] + query[(r & 63) * 512 + nn];
        pacc[mt][j] += tanhf(kv) * wv;
      }
    }
  }
#pragma unroll
  for (int mt = 0; mt < 2; ++mt)
#pragma unroll
    for (int j = 0; j < 8; ++j)
      atomicAdd(&align_s[mt * 16 + mb + j], pacc[mt][j]);
  __syncthreads();
  if (tid < KA_ROWS) align_out[r0 + tid] = align_s[tid];
}

// ---------------------------------------------------------------------------
__global__ void softmax_s_kernel(const float* __restrict__ align,
                                 float* __restrict__ attn) {
  __shared__ float red[256];
  int b = blockIdx.x, tid = threadIdx.x;
  float m = -3.4e38f;
  for (int s = tid; s < 400; s += 256) m = fmaxf(m, align[s * 64 + b]);
  red[tid] = m; __syncthreads();
  for (int off = 128; off > 0; off >>= 1) {
    if (tid < off) red[tid] = fmaxf(red[tid], red[tid + off]);
    __syncthreads();
  }
  m = red[0]; __syncthreads();
  float sum = 0.f;
  for (int s = tid; s < 400; s += 256) {
    float e = __expf(align[s * 64 + b] - m);
    attn[s * 64 + b] = e;
    sum += e;
  }
  red[tid] = sum; __syncthreads();
  for (int off = 128; off > 0; off >>= 1) {
    if (tid < off) red[tid] += red[tid + off];
    __syncthreads();
  }
  float inv = 1.0f / red[0];
  __syncthreads();
  for (int s = tid; s < 400; s += 256) attn[s * 64 + b] *= inv;
}

// ctx[b,h] = sum_s attn[s,b] * enc[s,b,h]   (grid: 64 x 4, block 256)
// enc likely still L2-resident from keys_align (192MB L2, NT weights elsewhere).
__global__ void ctx_kernel(const float* __restrict__ attn,
                           const float* __restrict__ enc,
                           float* __restrict__ ctx) {
  int b = blockIdx.x;
  int h = blockIdx.y * 256 + threadIdx.x;
  float acc = 0.f;
  for (int s = 0; s < 400; ++s)
    acc += attn[s * 64 + b] * enc[((size_t)s * 64 + b) * 1024 + h];
  ctx[b * 1024 + h] = acc;
}

// LSTM cell elementwise (gate order i,f,g,o). 65536 elems.
__global__ void lstm_kernel(const float* __restrict__ gates,
                            const float* __restrict__ c0,
                            float* __restrict__ h_out, float* __restrict__ c_out) {
  int idx = blockIdx.x * 256 + threadIdx.x;
  int b = idx >> 10, hh = idx & 1023;
  const float* g = gates + (size_t)b * 4096;
  float gi = g[hh], gf = g[1024 + hh], gg = g[2048 + hh], go = g[3072 + hh];
  float si = 1.f / (1.f + __expf(-gi));
  float sf = 1.f / (1.f + __expf(-gf));
  float so = 1.f / (1.f + __expf(-go));
  float c = sf * c0[idx] + si * tanhf(gg);
  float h = so * tanhf(c);
  c_out[idx] = c;
  h_out[idx] = h;
}

// ---- prep kernels ----------------------------------------------------------
__global__ void prep_xdec_kernel(const int* __restrict__ ids,
                                 const float* __restrict__ emb,
                                 const float* __restrict__ topics,
                                 float* __restrict__ xdec) {
  int b = blockIdx.x;
  const float* row = emb + (size_t)ids[b] * 512;
  for (int e = threadIdx.x; e < 523; e += 256)
    xdec[b * 523 + e] = (e < 512) ? row[e] : topics[b * 11 + (e - 512)];
}

__global__ void prep_x2_kernel(const float* __restrict__ feed_in,
                               float* __restrict__ x2) {
  int idx = blockIdx.x * 256 + threadIdx.x;  // 65536
  int b = idx >> 10, h = idx & 1023;
  x2[b * 1536 + 512 + h] = feed_in[idx];
}

__global__ void prep_xf_kernel(const float* __restrict__ h,
                               const float* __restrict__ ctx,
                               float* __restrict__ xf) {
  int idx = blockIdx.x * 256 + threadIdx.x;  // 131072
  int b = idx >> 11, k = idx & 2047;
  xf[idx] = (k < 1024) ? h[b * 1024 + k] : ctx[b * 1024 + (k - 1024)];
}

// W_attnm fp32 -> bf16 once (1MB; halves L2 traffic in the hot kernel)
__global__ void prep_wmb_kernel(const float* __restrict__ Wm,
                                bf16* __restrict__ Wmb) {
  int idx = blockIdx.x * 256 + threadIdx.x;  // 524288
  Wmb[idx] = (bf16)Wm[idx];
}

// ---------------------------------------------------------------------------
extern "C" void kernel_launch(void* const* d_in, const int* in_sizes, int n_in,
                              void* d_out, int out_size, void* d_ws, size_t ws_size,
                              hipStream_t stream) {
  const int*   ids     = (const int*)  d_in[0];
  const float* topics  = (const float*)d_in[1];
  const float* h0      = (const float*)d_in[2];
  const float* c0      = (const float*)d_in[3];
  const float* enc     = (const float*)d_in[4];
  const float* ifeed   = (const float*)d_in[5];
  const float* embt    = (const float*)d_in[6];
  const float* W_dec   = (const float*)d_in[7];
  const float* b_dec   = (const float*)d_in[8];
  const float* W_ih    = (const float*)d_in[9];
  const float* W_hh    = (const float*)d_in[10];
  const float* b_ih    = (const float*)d_in[11];
  const float* b_hh    = (const float*)d_in[12];
  const float* W_attnm = (const float*)d_in[13];
  const float* W_attnq = (const float*)d_in[14];
  const float* b_attnq = (const float*)d_in[15];
  const float* W_align = (const float*)d_in[16];
  const float* W_lin   = (const float*)d_in[17];
  const float* b_lin   = (const float*)d_in[18];
  const float* W_out   = (const float*)d_in[19];
  const float* b_out   = (const float*)d_in[20];

  // Output tuple flat: logits[1,64,V], h[1,64,H], c[1,64,H], attn[S,64], feed[1,64,H]
  float* out    = (float*)d_out;
  float* logits = out;
  float* h_out  = out + 3216448;            // 64*50257
  float* c_out  = out + 3281984;
  float* attn   = out + 3347520;
  float* feed   = out + 3373120;

  float* ws    = (float*)d_ws;
  float* xdec  = ws;                        // 64*523
  float* x2    = ws + 33792;                // 64*1536 (cols 0..511 = embedded)
  float* gates = ws + 132096;               // 64*4096
  float* query = ws + 394240;               // 64*512
  float* alg   = ws + 427008;               // 25600
  float* ctxb  = ws + 452864;               // 64*1024
  float* xf    = ws + 518400;               // 64*2048
  bf16*  wmb   = (bf16*)(ws + 649472);      // 512*1024 bf16

  prep_xdec_kernel<<<64, 256, 0, stream>>>(ids, embt, topics, xdec);
  prep_wmb_kernel<<<2048, 256, 0, stream>>>(W_attnm, wmb);
  // embedded = xdec @ W_dec^T + b_dec  -> into x2 cols 0..511 (ldc=1536)
  gemm_xwT_kernel<<<8, 128, 0, stream>>>(xdec, 523, W_dec, 523, b_dec,
                                         x2, 1536, 512, 523, 0);
  prep_x2_kernel<<<256, 256, 0, stream>>>(ifeed, x2);
  // gates = x2 @ W_ih^T + b_ih ; then += h0 @ W_hh^T + b_hh
  gemm_xwT_kernel<<<64, 128, 0, stream>>>(x2, 1536, W_ih, 1536, b_ih,
                                          gates, 4096, 4096, 1536, 0);
  gemm_xwT_kernel<<<64, 128, 0, stream>>>(h0, 1024, W_hh, 1024, b_hh,
                                          gates, 4096, 4096, 1024, 1);
  lstm_kernel<<<256, 256, 0, stream>>>(gates, c0, h_out, c_out);
  // query = h @ W_attnq^T + b_attnq
  gemm_xwT_kernel<<<8, 128, 0, stream>>>(h_out, 1024, W_attnq, 1024, b_attnq,
                                         query, 512, 512, 1024, 0);
  // fused keys->tanh->align (dominant GEMM; enc streamed once from HBM)
  keys_align_kernel<<<800, 256, 0, stream>>>(enc, wmb, query, W_align, alg);
  softmax_s_kernel<<<64, 256, 0, stream>>>(alg, attn);
  ctx_kernel<<<dim3(64, 4), 256, 0, stream>>>(attn, enc, ctxb);
  prep_xf_kernel<<<512, 256, 0, stream>>>(h_out, ctxb, xf);
  // feed = [h,ctx] @ W_lin^T + b_lin  (straight into d_out feed slot)
  gemm_xwT_kernel<<<16, 128, 0, stream>>>(xf, 2048, W_lin, 2048, b_lin,
                                          feed, 1024, 1024, 2048, 0);
  // logits = feed @ W_out^T + b_out  (N=50257; W_out streamed exactly once, NT)
  gemm_xwT_kernel<<<786, 128, 0, stream>>>(feed, 1024, W_out, 1024, b_out,
                                           logits, 50257, 50257, 1024, 0);
}
